// LFE_62612033241312
// MI455X (gfx1250) — compile-verified, matrix operand order checked by reading, not run on hardware
//
#include <hip/hip_runtime.h>
#include <hip/hip_bf16.h>

typedef __attribute__((ext_vector_type(16))) __bf16 v16bf;
typedef __attribute__((ext_vector_type(8)))  __bf16 v8bf;
typedef __attribute__((ext_vector_type(8)))  float  v8f;

#define EPS 1e-5f

#define BATCH 4
#define CCH   256
#define HH    128
#define WW    128
#define HWSZ  16384
#define PLANE (CCH*HWSZ)
#define ANUM  128
#define RNUM  32
#define PADW  130            // 128 + halo

__device__ __forceinline__ v8f wmma_bf16(v16bf a, v16bf b, v8f c) {
  return __builtin_amdgcn_wmma_f32_16x16x32_bf16(false, a, false, b, (short)0, c,
                                                 false, false);
}

// fragment = two contiguous 8-element (16B) runs: k = c0+half*8+[0,8) and +16
__device__ __forceinline__ v16bf load_frag(const __bf16* p) {
  const v8bf lo = *(const v8bf*)p;
  const v8bf hi = *(const v8bf*)(p + 16);
  v16bf r;
#pragma unroll
  for (int j = 0; j < 8; ++j) { r[j] = lo[j]; r[j + 8] = hi[j]; }
  return r;
}

__device__ __forceinline__ float bn_apply(float v, const float* bn, int c, int C) {
  float g = bn[c], be = bn[C + c], mn = bn[2 * C + c], vr = bn[3 * C + c];
  float inv = g * rsqrtf(vr + EPS);
  return v * inv + (be - mn * inv);
}

// ========================= pre-pass kernels =========================
__global__ __launch_bounds__(256) void k_zero_u32(unsigned int* p, long n) {
  long i = (long)blockIdx.x * 256 + threadIdx.x;
  long stride = (long)gridDim.x * 256;
  for (; i < n; i += stride) p[i] = 0u;
}

// x fp32 NCHW -> bf16 padded NHWC  xp[b][y+1][x+1][c]
__global__ __launch_bounds__(256) void k_pack_x(const float* __restrict__ x,
                                                __bf16* __restrict__ xp) {
  __shared__ float t[64][65];
  const int b = blockIdx.z, y = blockIdx.y;
  const int xt = blockIdx.x & 1, ct = blockIdx.x >> 1;
  const int tid = threadIdx.x;
  const int rx = tid & 63, rc = tid >> 6;
  for (int cc = rc; cc < 64; cc += 4)
    t[cc][rx] = x[((size_t)(b * CCH + ct * 64 + cc) * HH + y) * WW + xt * 64 + rx];
  __syncthreads();
  const int wc = tid & 63, wx = tid >> 6;
  for (int xx = wx; xx < 64; xx += 4)
    xp[(((size_t)b * PADW + y + 1) * PADW + xt * 64 + xx + 1) * 256 + ct * 64 + wc] =
        (__bf16)t[wc][xx];
}

// conv weights [kout][c][tap] fp32 -> bf16 [kout(512)][tap(9)][c(256)]
__global__ __launch_bounds__(256) void k_pack_w(const float* __restrict__ w1,
                                                const float* __restrict__ w2,
                                                __bf16* __restrict__ wpk) {
  const int kout = blockIdx.x;
  const float* src = ((kout < 256) ? w1 : w2) + (size_t)(kout & 255) * 2304;
  const int c = threadIdx.x;
  for (int tap = 0; tap < 9; ++tap)
    wpk[((size_t)kout * 9 + tap) * 256 + c] = (__bf16)src[c * 9 + tap];
}

__global__ __launch_bounds__(256) void k_cast_bf(const float* __restrict__ s,
                                                 __bf16* __restrict__ d, int n) {
  int i = blockIdx.x * 256 + threadIdx.x;
  if (i < n) d[i] = (__bf16)s[i];
}

// =====================================================================
// K1: conv1+conv2 (3x3, pad1) implicit GEMM, M=512 concat out-channels.
// B from zero-padded bf16 NHWC -> no predication, b128 fragment loads.
// Epilogue -> fbuf[b][p][512] bf16 (fore: 0..255, back: 256..511).
// grid(256, 8, 4), block 256 (8 waves)
// =====================================================================
__global__ __launch_bounds__(256) void k_conv3x3_dual(
    const __bf16* __restrict__ xp, const __bf16* __restrict__ wpk,
    const float* __restrict__ bn1, const float* __restrict__ a1p,
    const float* __restrict__ bn2, const float* __restrict__ a2p,
    const float* __restrict__ bnr, const float* __restrict__ arp,
    __bf16* __restrict__ fbuf)
{
  const int lane = threadIdx.x & 31;
  const int wid  = threadIdx.x >> 5;
  const int wm = wid & 3, wn = wid >> 2;
  const int half = lane >> 4, l15 = lane & 15;
  const int b = blockIdx.z;
  const int mbase = blockIdx.y * 64 + wm * 16;
  const int tile = blockIdx.x;
  const int y0 = tile >> 1;
  const int xb = (tile & 1) * 64 + wn * 32;

  const int kout = mbase + l15;
  const __bf16* wrow = wpk + (size_t)kout * 9 * 256 + half * 8;

  v8f acc0 = {}; v8f acc1 = {};
  for (int tap = 0; tap < 9; ++tap) {
    const int dy = tap / 3 - 1, dx = tap % 3 - 1;
    const size_t rb =
        (((size_t)b * PADW + (y0 + dy + 1)) * PADW + (xb + l15 + dx + 1)) * 256;
    const __bf16* arow = wrow + tap * 256;
    const __bf16* brow0 = xp + rb + half * 8;
    const __bf16* brow1 = xp + rb + 16 * 256 + half * 8;   // +16 pixels
    for (int ks = 0; ks < 8; ++ks) {
      const int c0 = ks * 32;
      __builtin_prefetch(brow0 + c0 + 32, 0, 1);
      __builtin_prefetch(brow1 + c0 + 32, 0, 1);
      v16bf af = load_frag(arow + c0);
      v16bf bf0 = load_frag(brow0 + c0);
      v16bf bf1 = load_frag(brow1 + c0);
      acc0 = wmma_bf16(af, bf0, acc0);
      acc1 = wmma_bf16(af, bf1, acc1);
    }
  }

  const float a1 = a1p[0], a2 = a2p[0], ar = arp[0];
  const int chb = mbase + half * 8;            // 8 contiguous channels
#pragma unroll
  for (int nf = 0; nf < 2; ++nf) {
    const int p = y0 * WW + xb + nf * 16 + l15;
    __bf16* dst = fbuf + ((size_t)b * HWSZ + p) * 512 + chb;
    v8bf o;
#pragma unroll
    for (int r = 0; r < 8; ++r) {
      const int ch = chb + r;
      const float v = nf ? acc1[r] : acc0[r];
      float res;
      if (ch < 256) {
        float yv = bn_apply(v, bn1, ch, 256);
        res = yv > 0.f ? yv : a1 * yv;
      } else {
        const int c2 = ch - 256;
        float t = bn_apply(v, bn2, c2, 256);
        t = t > 0.f ? t : a2 * t;
        float t2 = bn_apply(1.f - t, bnr, c2, 256);
        res = t2 > 0.f ? t2 : ar * t2;
      }
      o[r] = (__bf16)res;
    }
    *(v8bf*)dst = o;
  }
}

// =====================================================================
// K2: cat 1x1 (512->256) + BN + PReLU -> x12 bf16 NHWC.
// grid(256, 4, 4), block 256
// =====================================================================
__global__ __launch_bounds__(256) void k_cat1x1(
    const __bf16* __restrict__ fbuf, const __bf16* __restrict__ catw,
    const float* __restrict__ bnc, const float* __restrict__ acp,
    __bf16* __restrict__ x12)
{
  const int lane = threadIdx.x & 31;
  const int wid  = threadIdx.x >> 5;
  const int wm = wid & 3, wn = wid >> 2;
  const int half = lane >> 4, l15 = lane & 15;
  const int b = blockIdx.z;
  const int mbase = blockIdx.y * 64 + wm * 16;
  const int p0 = blockIdx.x * 64 + wn * 32;
  const __bf16* arow = catw + (size_t)(mbase + l15) * 512 + half * 8;
  const __bf16* brow0 = fbuf + ((size_t)b * HWSZ + p0 + l15) * 512 + half * 8;
  const __bf16* brow1 = fbuf + ((size_t)b * HWSZ + p0 + 16 + l15) * 512 + half * 8;

  v8f acc0 = {}; v8f acc1 = {};
  for (int ks = 0; ks < 16; ++ks) {
    const int c0 = ks * 32;
    v16bf af = load_frag(arow + c0);
    v16bf bf0 = load_frag(brow0 + c0);
    v16bf bf1 = load_frag(brow1 + c0);
    acc0 = wmma_bf16(af, bf0, acc0);
    acc1 = wmma_bf16(af, bf1, acc1);
  }

  const float aa = acp[0];
  const int chb = mbase + half * 8;
#pragma unroll
  for (int nf = 0; nf < 2; ++nf) {
    const int p = p0 + nf * 16 + l15;
    __bf16* dst = x12 + ((size_t)b * HWSZ + p) * 256 + chb;
    v8bf o;
#pragma unroll
    for (int r = 0; r < 8; ++r) {
      float yv = bn_apply(nf ? acc1[r] : acc0[r], bnc, chb + r, 256);
      o[r] = (__bf16)(yv > 0.f ? yv : aa * yv);
    }
    *(v8bf*)dst = o;
  }
}

// =====================================================================
// K3: attc(256->2)+BN+PReLU+softmax, mid = fore*a0+back*a1 (in-place x12)
// grid(64, 4), block 256
// =====================================================================
__global__ __launch_bounds__(256) void k_att_mid(
    __bf16* __restrict__ x12, const __bf16* __restrict__ fbuf,
    const float* __restrict__ attw, const float* __restrict__ bna,
    const float* __restrict__ aap)
{
  const int b = blockIdx.y;
  const int p = blockIdx.x * 256 + threadIdx.x;
  const __bf16* xr = x12 + ((size_t)b * HWSZ + p) * 256;
  float z0 = 0.f, z1 = 0.f;
  for (int c = 0; c < 256; ++c) {
    float v = (float)xr[c];
    z0 = fmaf(attw[c], v, z0);
    z1 = fmaf(attw[256 + c], v, z1);
  }
  const float aa = aap[0];
  float inv0 = bna[0] * rsqrtf(bna[6] + EPS);
  float inv1 = bna[1] * rsqrtf(bna[7] + EPS);
  float z0b = z0 * inv0 + (bna[2] - bna[4] * inv0);
  float z1b = z1 * inv1 + (bna[3] - bna[5] * inv1);
  z0b = z0b > 0.f ? z0b : aa * z0b;
  z1b = z1b > 0.f ? z1b : aa * z1b;
  float mx = fmaxf(z0b, z1b);
  float e0 = __expf(z0b - mx), e1 = __expf(z1b - mx);
  float at0 = e0 / (e0 + e1), at1 = 1.f - at0;
  const __bf16* fr = fbuf + ((size_t)b * HWSZ + p) * 512;
  __bf16* mr = x12 + ((size_t)b * HWSZ + p) * 256;
  for (int c = 0; c < 256; ++c)
    mr[c] = (__bf16)((float)fr[c] * at0 + (float)fr[256 + c] * at1);
}

// =====================================================================
// K4: smooth 1x1 (256->256) + BN + PReLU -> out (fp32 NCHW, d_out)
// grid(256, 4, 4), block 256
// =====================================================================
__global__ __launch_bounds__(256) void k_smooth1x1(
    const __bf16* __restrict__ mid, const __bf16* __restrict__ sw,
    const float* __restrict__ bns, const float* __restrict__ asp,
    float* __restrict__ outp)
{
  const int lane = threadIdx.x & 31;
  const int wid  = threadIdx.x >> 5;
  const int wm = wid & 3, wn = wid >> 2;
  const int half = lane >> 4, l15 = lane & 15;
  const int b = blockIdx.z;
  const int mbase = blockIdx.y * 64 + wm * 16;
  const int p0 = blockIdx.x * 64 + wn * 32;
  const __bf16* arow = sw + (size_t)(mbase + l15) * 256 + half * 8;
  const __bf16* brow0 = mid + ((size_t)b * HWSZ + p0 + l15) * 256 + half * 8;
  const __bf16* brow1 = mid + ((size_t)b * HWSZ + p0 + 16 + l15) * 256 + half * 8;

  v8f acc0 = {}; v8f acc1 = {};
  for (int ks = 0; ks < 8; ++ks) {
    const int c0 = ks * 32;
    v16bf af = load_frag(arow + c0);
    v16bf bf0 = load_frag(brow0 + c0);
    v16bf bf1 = load_frag(brow1 + c0);
    acc0 = wmma_bf16(af, bf0, acc0);
    acc1 = wmma_bf16(af, bf1, acc1);
  }

  const float aa = asp[0];
#pragma unroll
  for (int r = 0; r < 8; ++r) {
    const int ch = mbase + half * 8 + r;
#pragma unroll
    for (int nf = 0; nf < 2; ++nf) {
      float yv = bn_apply(nf ? acc1[r] : acc0[r], bns, ch, 256);
      yv = yv > 0.f ? yv : aa * yv;
      outp[(size_t)(b * CCH + ch) * HWSZ + p0 + nf * 16 + l15] = yv;
    }
  }
}

// =====================================================================
// K5: final 1x1 (256->1)+bias -> o1 ; ht1 scalar conv+BN+ReLU -> h image
// grid(64, 4), block 256
// =====================================================================
__global__ __launch_bounds__(256) void k_final_ht1(
    const float* __restrict__ outp, const float* __restrict__ fw,
    const float* __restrict__ fb, const float* __restrict__ h1w,
    const float* __restrict__ h1b, const float* __restrict__ h1bn,
    float* __restrict__ o1, float* __restrict__ himg)
{
  const int b = blockIdx.y;
  const int p = blockIdx.x * 256 + threadIdx.x;
  const float* oc = outp + (size_t)b * PLANE + p;
  float s = 0.f;
  for (int c = 0; c < 256; ++c) s = fmaf(fw[c], oc[(size_t)c * HWSZ], s);
  s += fb[0];
  o1[b * HWSZ + p] = s;
  float hv = s * h1w[0] + h1b[0];
  float inv = h1bn[0] * rsqrtf(h1bn[3] + EPS);
  float y = hv * inv + (h1bn[1] - h1bn[2] * inv);
  himg[b * HWSZ + p] = fmaxf(y, 0.f);
}

// =====================================================================
// K6: Hough scatter, image in LDS, per-(wave,angle) LDS accumulators.
// grid(8, 4), block 256
// =====================================================================
__global__ __launch_bounds__(256) void k_dht(
    const float* __restrict__ himg, float* __restrict__ hacc)
{
  __shared__ float simg[HWSZ];
  __shared__ float sacc[16][RNUM];
  const int b = blockIdx.y;
  const int grp = blockIdx.x;
  const int tid = threadIdx.x;
  const int lane = tid & 31, wid = tid >> 5;
  for (int i = tid; i < HWSZ; i += 256) simg[i] = himg[b * HWSZ + i];
  __syncthreads();
  const float inv_irho = 31.0f / 182.0f;
  for (int aa = 0; aa < 2; ++aa) {
    const int ang = grp * 16 + wid * 2 + aa;
    const float th = (float)ang * (3.14159265358979323846f / 128.f);
    float st, ct;
    __sincosf(th, &st, &ct);
    float* acc = sacc[wid * 2 + aa];
    acc[lane] = 0.f;
    for (int i = lane; i < HWSZ; i += 32) {
      const float v = simg[i];
      const int yv = i >> 7, xw = i & 127;
      const float r = (ct * (float)(xw - 64) + st * (float)(yv - 64)) * inv_irho;
      int idx = (int)rintf(r) + RNUM / 2;
      idx = min(RNUM - 1, max(0, idx));
      atomicAdd(&acc[idx], v);
    }
    hacc[((size_t)b * ANUM + ang) * RNUM + lane] = acc[lane];
  }
}

// =====================================================================
// K7: ht2 + ht3 (3x3, 1ch) on [128,32] in LDS. grid(4), block 256
// =====================================================================
__global__ __launch_bounds__(256) void k_ht23(
    const float* __restrict__ hacc,
    const float* __restrict__ w2, const float* __restrict__ b2,
    const float* __restrict__ bn2p, const float* __restrict__ w3,
    const float* __restrict__ b3, const float* __restrict__ bn3p,
    float* __restrict__ hout)
{
  __shared__ float s0[ANUM * RNUM];
  __shared__ float s1[ANUM * RNUM];
  const int b = blockIdx.x;
  const int tid = threadIdx.x;
  for (int i = tid; i < ANUM * RNUM; i += 256) s0[i] = hacc[b * ANUM * RNUM + i];
  __syncthreads();
  const float inv2 = bn2p[0] * rsqrtf(bn2p[3] + EPS);
  const float sh2 = bn2p[1] - bn2p[2] * inv2;
  for (int i = tid; i < ANUM * RNUM; i += 256) {
    const int y = i >> 5, xw = i & 31;
    float s = b2[0];
#pragma unroll
    for (int t = 0; t < 9; ++t) {
      const int dy = t / 3 - 1, dx = t % 3 - 1;
      const int yy = y + dy, xx = xw + dx;
      if ((unsigned)yy < (unsigned)ANUM && (unsigned)xx < (unsigned)RNUM)
        s = fmaf(w2[t], s0[yy * RNUM + xx], s);
    }
    s1[i] = fmaxf(s * inv2 + sh2, 0.f);
  }
  __syncthreads();
  const float inv3 = bn3p[0] * rsqrtf(bn3p[3] + EPS);
  const float sh3 = bn3p[1] - bn3p[2] * inv3;
  for (int i = tid; i < ANUM * RNUM; i += 256) {
    const int y = i >> 5, xw = i & 31;
    float s = b3[0];
#pragma unroll
    for (int t = 0; t < 9; ++t) {
      const int dy = t / 3 - 1, dx = t % 3 - 1;
      const int yy = y + dy, xx = xw + dx;
      if ((unsigned)yy < (unsigned)ANUM && (unsigned)xx < (unsigned)RNUM)
        s = fmaf(w3[t], s1[yy * RNUM + xx], s);
    }
    hout[b * ANUM * RNUM + i] = fmaxf(s * inv3 + sh3, 0.f);
  }
}

// =====================================================================
extern "C" void kernel_launch(void* const* d_in, const int* in_sizes, int n_in,
                              void* d_out, int out_size, void* d_ws, size_t ws_size,
                              hipStream_t stream) {
  (void)in_sizes; (void)n_in; (void)out_size; (void)ws_size;
  const float* x        = (const float*)d_in[0];
  const float* conv1_w  = (const float*)d_in[1];
  const float* conv1_bn = (const float*)d_in[2];
  const float* conv1_a  = (const float*)d_in[3];
  const float* conv2_w  = (const float*)d_in[4];
  const float* conv2_bn = (const float*)d_in[5];
  const float* conv2_a  = (const float*)d_in[6];
  const float* bnr_bn   = (const float*)d_in[7];
  const float* bnr_a    = (const float*)d_in[8];
  const float* cat_w    = (const float*)d_in[9];
  const float* cat_bn   = (const float*)d_in[10];
  const float* cat_a    = (const float*)d_in[11];
  const float* attc_w   = (const float*)d_in[12];
  const float* attc_bn  = (const float*)d_in[13];
  const float* attc_a   = (const float*)d_in[14];
  const float* smooth_w = (const float*)d_in[15];
  const float* smooth_bn= (const float*)d_in[16];
  const float* smooth_a = (const float*)d_in[17];
  const float* final_w  = (const float*)d_in[18];
  const float* final_b  = (const float*)d_in[19];
  const float* ht1_w    = (const float*)d_in[20];
  const float* ht1_b    = (const float*)d_in[21];
  const float* ht1_bn   = (const float*)d_in[22];
  const float* ht2_w    = (const float*)d_in[23];
  const float* ht2_b    = (const float*)d_in[24];
  const float* ht2_bn   = (const float*)d_in[25];
  const float* ht3_w    = (const float*)d_in[26];
  const float* ht3_b    = (const float*)d_in[27];
  const float* ht3_bn   = (const float*)d_in[28];

  float* outp = (float*)d_out;                        // [4,256,128,128]
  float* o1   = outp + (size_t)BATCH * PLANE;         // [4,1,128,128]
  float* hout = o1 + BATCH * HWSZ;                    // [4,1,128,32]

  // ---- workspace carve-up ----
  char* w = (char*)d_ws;
  const size_t xp_n   = (size_t)BATCH * PADW * PADW * 256;   // bf16
  const size_t wpk_n  = (size_t)512 * 9 * 256;               // bf16
  const size_t catw_n = 256 * 512;                           // bf16
  const size_t sw_n   = 256 * 256;                           // bf16
  const size_t fbuf_n = (size_t)BATCH * HWSZ * 512;          // bf16
  const size_t x12_n  = (size_t)BATCH * HWSZ * 256;          // bf16

  __bf16* xp    = (__bf16*)w;                 w += xp_n * 2;
  __bf16* wpk   = (__bf16*)w;                 w += wpk_n * 2;
  __bf16* catwb = (__bf16*)w;                 w += catw_n * 2;
  __bf16* swb   = (__bf16*)w;                 w += sw_n * 2;
  __bf16* fbuf  = (__bf16*)w;                 w += fbuf_n * 2;
  __bf16* x12   = (__bf16*)w;                 w += x12_n * 2;
  float*  himg  = (float*)w;                  w += (size_t)BATCH * HWSZ * 4;
  float*  hacc  = (float*)w;

  dim3 blk(256);
  // pre-passes
  long zwords = (long)(xp_n / 2);            // bf16 pairs as u32
  k_zero_u32<<<dim3(4096), blk, 0, stream>>>((unsigned int*)xp, zwords);
  k_pack_x<<<dim3(8, 128, 4), blk, 0, stream>>>(x, xp);
  k_pack_w<<<dim3(512), blk, 0, stream>>>(conv1_w, conv2_w, wpk);
  k_cast_bf<<<dim3((int)((catw_n + 255) / 256)), blk, 0, stream>>>(cat_w, catwb, (int)catw_n);
  k_cast_bf<<<dim3((int)((sw_n + 255) / 256)), blk, 0, stream>>>(smooth_w, swb, (int)sw_n);

  // main pipeline
  k_conv3x3_dual<<<dim3(256, 8, 4), blk, 0, stream>>>(
      xp, wpk, conv1_bn, conv1_a, conv2_bn, conv2_a, bnr_bn, bnr_a, fbuf);
  k_cat1x1<<<dim3(256, 4, 4), blk, 0, stream>>>(fbuf, catwb, cat_bn, cat_a, x12);
  k_att_mid<<<dim3(64, 4), blk, 0, stream>>>(x12, fbuf, attc_w, attc_bn, attc_a);
  k_smooth1x1<<<dim3(256, 4, 4), blk, 0, stream>>>(x12, swb, smooth_bn, smooth_a, outp);
  k_final_ht1<<<dim3(64, 4), blk, 0, stream>>>(outp, final_w, final_b, ht1_w, ht1_b,
                                               ht1_bn, o1, himg);
  k_dht<<<dim3(8, 4), blk, 0, stream>>>(himg, hacc);
  k_ht23<<<dim3(4), blk, 0, stream>>>(hacc, ht2_w, ht2_b, ht2_bn, ht3_w, ht3_b,
                                      ht3_bn, hout);
}
